// Qwen3Attention_79482664779984
// MI455X (gfx1250) — matrix operand order repaired and checked by reading, not compiled, not run on hardware
//
#include <hip/hip_runtime.h>
#include <hip/hip_bf16.h>

// ---------- types ----------
typedef __attribute__((ext_vector_type(16))) __bf16 v16bf;
typedef __attribute__((ext_vector_type(8)))  float  v8f;
typedef __attribute__((ext_vector_type(4)))  unsigned int u32x4;
typedef int v4i __attribute__((vector_size(16)));   // matches async-LDS builtin proto

union Frag {                 // one WMMA A/B operand: 16 bf16 = 32 bytes
    v16bf v;
    u32x4 q[2];
    unsigned short us[16];
};

__device__ __forceinline__ unsigned short f2bf(float f) {
    unsigned u = __builtin_bit_cast(unsigned, f);
    u += 0x7FFFu + ((u >> 16) & 1u);           // round-to-nearest-even
    return (unsigned short)(u >> 16);
}

// ---------- CDNA5 async global->LDS staging (ASYNCcnt path) ----------
#if __has_builtin(__builtin_amdgcn_global_load_async_to_lds_b128)
#define HAVE_ASYNC_LDS 1
__device__ __forceinline__ void async_copy16(const void* g, void* l) {
    __builtin_amdgcn_global_load_async_to_lds_b128(
        (__attribute__((address_space(1))) v4i*)g,
        (__attribute__((address_space(3))) v4i*)l, 0, 0);
}
#else
#define HAVE_ASYNC_LDS 0
__device__ __forceinline__ void async_copy16(const void* g, void* l) {
    *(u32x4*)l = *(const u32x4*)g;             // fallback: through VGPRs
}
#endif

__device__ __forceinline__ void wait_async_lds() {
#if HAVE_ASYNC_LDS && __has_builtin(__builtin_amdgcn_s_wait_asynccnt)
    __builtin_amdgcn_s_wait_asynccnt(0);
#endif
}

// ---------- problem constants ----------
#define BB   2
#define SEQ  2048
#define DMODEL 2048
#define NH   16
#define NKVH 8
#define HD   128
#define MROWS (BB*SEQ)          // 4096
#define ATT_SCALE 0.08838834764831845f   // 1/sqrt(128)

// =====================================================================
// cast fp32 -> bf16 (flat)
// =====================================================================
__global__ __launch_bounds__(256) void cast_bf16(const float* __restrict__ in,
                                                 unsigned short* __restrict__ out,
                                                 long long n) {
    long long i = (long long)blockIdx.x * 256 + threadIdx.x;
    if (i < n) out[i] = f2bf(in[i]);
}

// pack W[N][K] fp32 -> Wt[K][N] bf16
__global__ __launch_bounds__(256) void pack_wT(const float* __restrict__ W,
                                               unsigned short* __restrict__ Wt,
                                               int N, int K) {
    long long i = (long long)blockIdx.x * 256 + threadIdx.x;   // i = k*N + n
    if (i >= (long long)N * K) return;
    int n = (int)(i % N);
    int k = (int)(i / N);
    Wt[i] = f2bf(W[(long long)n * K + k]);
}

// V fp32 [B,S,KVH*128] -> bf16 [B,KVH,128,S]  (dim-major for P@V B-operand)
__global__ __launch_bounds__(256) void v_trans(const float* __restrict__ V,
                                               unsigned short* __restrict__ Vt) {
    long long i = (long long)blockIdx.x * 256 + threadIdx.x;
    if (i >= (long long)BB * NKVH * HD * SEQ) return;
    int s  = (int)(i % SEQ);
    long long r = i / SEQ;
    int d  = (int)(r % HD);   r /= HD;
    int kv = (int)(r % NKVH);
    int b  = (int)(r / NKVH);
    Vt[i] = f2bf(V[(((long long)b * SEQ + s) * NKVH + kv) * HD + d]);
}

// =====================================================================
// bf16 WMMA GEMM:  C[M,N] (fp32) = A[M,K](bf16) * B[K,N](bf16)
// 128x128 tile / workgroup, 8 waves in 4(M)x2(N) grid, each wave 32x64.
// A tile staged via GLOBAL_LOAD_ASYNC_TO_LDS; B tile transposed via lanes.
// =====================================================================
__global__ __launch_bounds__(256) void gemm_bf16(const unsigned short* __restrict__ A,
                                                 const unsigned short* __restrict__ Bm,
                                                 float* __restrict__ C,
                                                 int M, int N, int K) {
    __shared__ unsigned short sA[128 * 32];   // [m][k]
    __shared__ unsigned short sB[128 * 32];   // [n][k]  (transposed in LDS)

    const int tn = blockIdx.x, tm = blockIdx.y;
    const int tid = threadIdx.x;
    const int wave = tid >> 5, lane = tid & 31;
    const int l16 = lane & 15, hi = lane >> 4;
    const int wm = wave & 3;         // 0..3  -> 32-row slab
    const int wn = wave >> 2;        // 0..1  -> 64-col slab

    v8f acc[2][4];
    const v8f vz = {0.f,0.f,0.f,0.f,0.f,0.f,0.f,0.f};
    for (int i = 0; i < 2; ++i) for (int j = 0; j < 4; ++j) acc[i][j] = vz;

    for (int kk = 0; kk < K; kk += 32) {
        // ---- stage A tile [128][32] asynchronously (no VGPR round trip) ----
        {
            int row = tid >> 1;
            int kh  = (tid & 1) * 16;
            const unsigned short* src = A + (long long)(tm*128 + row) * K + kk + kh;
            unsigned short* dst = sA + row * 32 + kh;
            async_copy16(src,     dst);
            async_copy16(src + 8, dst + 8);
        }
        // ---- stage B tile transposed -> [n][k] (must pass through lanes) ----
        {
            int kr = tid >> 3;              // 0..31
            int n0 = (tid & 7) * 16;        // 0..112
            Frag f;
            const unsigned short* src = Bm + (long long)(kk + kr) * N + tn*128 + n0;
            f.q[0] = *(const u32x4*)(src);
            f.q[1] = *(const u32x4*)(src + 8);
            for (int i = 0; i < 16; ++i) sB[(n0 + i) * 32 + kr] = f.us[i];
        }
        // prefetch next K-slice into near cache while we compute
        if (kk + 32 < K) {
            __builtin_prefetch(A + (long long)(tm*128 + (tid >> 1)) * K + kk + 32, 0, 3);
            __builtin_prefetch(Bm + (long long)(kk + 32 + (tid >> 3)) * N + tn*128, 0, 3);
        }
        wait_async_lds();
        __syncthreads();

        // ---- WMMA: 2 (M) x 4 (N) subtiles, single K-step of 32 ----
        Frag aF[2];
        const int kb = hi ? 8 : 0;
        for (int i = 0; i < 2; ++i) {
            int row = wm*32 + i*16 + l16;
            aF[i].q[0] = *(const u32x4*)(sA + row*32 + kb);
            aF[i].q[1] = *(const u32x4*)(sA + row*32 + 16 + kb);
        }
        const int kb2 = hi ? 8 : 0;
        for (int j = 0; j < 4; ++j) {
            Frag bF;
            int n = wn*64 + j*16 + l16;
            bF.q[0] = *(const u32x4*)(sB + n*32 + kb2);
            bF.q[1] = *(const u32x4*)(sB + n*32 + 16 + kb2);
            for (int i = 0; i < 2; ++i)
                acc[i][j] = __builtin_amdgcn_wmma_f32_16x16x32_bf16(
                    false, aF[i].v, false, bF.v, (short)0, acc[i][j], false, false);
        }
        __syncthreads();
    }

    // ---- store C (D-layout: vgpr r -> row r / r+8, lane -> col) ----
    for (int i = 0; i < 2; ++i)
        for (int j = 0; j < 4; ++j)
            for (int r = 0; r < 8; ++r) {
                int row = tm*128 + wm*32 + i*16 + r + (hi ? 8 : 0);
                int col = tn*128 + wn*64 + j*16 + l16;
                C[(long long)row * N + col] = acc[i][j][r];
            }
}

// =====================================================================
// per-head RMSNorm + RoPE; write bf16 [B,H,S,128]
// 64 threads / block, one (b,s,h); thread t owns pair (2t, 2t+1)
// =====================================================================
__global__ __launch_bounds__(64) void norm_rope(const float* __restrict__ X,
                                                const float* __restrict__ fc,
                                                const float* __restrict__ w,
                                                unsigned short* __restrict__ Out,
                                                int nh) {
    int bid = blockIdx.x;
    int h = bid % nh;
    int s = (bid / nh) % SEQ;
    int b = bid / (nh * SEQ);
    int t = threadIdx.x;

    const float* xp = X + (((long long)b * SEQ + s) * nh + h) * HD;
    float x0 = xp[2*t], x1 = xp[2*t + 1];
    float ss = x0*x0 + x1*x1;
    for (int m = 16; m >= 1; m >>= 1) ss += __shfl_xor(ss, m, 32);
    __shared__ float red[2];
    if ((t & 31) == 0) red[t >> 5] = ss;
    __syncthreads();
    float inv = rsqrtf((red[0] + red[1]) * (1.0f / 128.0f) + 1e-6f);

    float n0 = x0 * inv * w[2*t];
    float n1 = x1 * inv * w[2*t + 1];
    float c = fc[(long long)s * HD + 2*t];
    float si = fc[(long long)s * HD + 2*t + 1];
    float r0 = n0 * c - n1 * si;
    float r1 = n0 * si + n1 * c;
    unsigned short* op = Out + (((long long)b * nh + h) * SEQ + s) * HD;
    op[2*t]     = f2bf(r0);
    op[2*t + 1] = f2bf(r1);
}

// =====================================================================
// Flash attention (causal, GQA rep=2).
// Block: (b, h, 128-query tile), 8 waves, each wave owns 16 query rows.
// LDS: K tile [key][dim] 32KB + V tile [dim][key] 32KB + P [8][16][128] 32KB
// K/V tiles staged via GLOBAL_LOAD_ASYNC_TO_LDS (ASYNCcnt).
// =====================================================================
__global__ __launch_bounds__(256) void flash_attn(const unsigned short* __restrict__ Q,
                                                  const unsigned short* __restrict__ Kn,
                                                  const unsigned short* __restrict__ Vt,
                                                  unsigned short* __restrict__ O) {
    extern __shared__ unsigned short sm[];
    unsigned short* sK = sm;                // 128*128
    unsigned short* sV = sm + 16384;        // 128*128
    unsigned short* sP = sm + 32768;        // 8*16*128

    const int QT = SEQ >> 7;                // 16 query tiles
    int qt = blockIdx.x % QT;
    int h  = (blockIdx.x / QT) % NH;
    int b  = blockIdx.x / (QT * NH);
    int kv = h >> 1;

    const int tid = threadIdx.x, wave = tid >> 5, lane = tid & 31;
    const int l16 = lane & 15, hi = lane >> 4;
    const int kb  = hi ? 8 : 0;     // A-operand K base
    const int kbB = hi ? 8 : 0;     // B-operand K base

    // Q fragments (A layout), 4 chunks of K=32 across head dim
    Frag qF[4];
    {
        int qrow = qt*128 + wave*16 + l16;
        const unsigned short* qp = Q + (((long long)(b*NH + h)) * SEQ + qrow) * HD;
        for (int c = 0; c < 4; ++c) {
            qF[c].q[0] = *(const u32x4*)(qp + c*32 + kb);
            qF[c].q[1] = *(const u32x4*)(qp + c*32 + 16 + kb);
        }
    }

    const v8f vz = {0.f,0.f,0.f,0.f,0.f,0.f,0.f,0.f};
    v8f o[8]; for (int d = 0; d < 8; ++d) o[d] = vz;
    float mrow[8], lrow[8];
    for (int r = 0; r < 8; ++r) { mrow[r] = -__builtin_inff(); lrow[r] = 0.f; }

    const unsigned short* kg = Kn + ((long long)(b*NKVH + kv)) * SEQ * HD;
    const unsigned short* vg = Vt + ((long long)(b*NKVH + kv)) * HD * SEQ;

    for (int kt = 0; kt <= qt; ++kt) {
        __syncthreads();
        // stage K tile [key][dim] and V tile [dim][key] asynchronously
        {
            int row  = tid >> 1;
            int half = (tid & 1) * 64;
            const unsigned short* s1 = kg + ((long long)(kt*128 + row)) * HD + half;
            unsigned short* d1 = sK + row*128 + half;
            const unsigned short* s2 = vg + (long long)row * SEQ + kt*128 + half;
            unsigned short* d2 = sV + row*128 + half;
            for (int i = 0; i < 8; ++i) {
                async_copy16(s1 + i*8, d1 + i*8);
                async_copy16(s2 + i*8, d2 + i*8);
            }
        }
        wait_async_lds();
        __syncthreads();

        // S = Q @ K^T : 8 key subtiles x 4 K-chunks
        v8f sc[8];
        for (int t = 0; t < 8; ++t) {
            v8f a = vz;
            for (int c = 0; c < 4; ++c) {
                Frag bF;
                int n = t*16 + l16;                 // key column
                bF.q[0] = *(const u32x4*)(sK + n*128 + c*32 + kbB);
                bF.q[1] = *(const u32x4*)(sK + n*128 + c*32 + 16 + kbB);
                a = __builtin_amdgcn_wmma_f32_16x16x32_bf16(
                        false, qF[c].v, false, bF.v, (short)0, a, false, false);
            }
            sc[t] = a;
        }

        // scale + causal mask (only the diagonal tile needs it)
        const bool diag = (kt == qt);
        for (int t = 0; t < 8; ++t)
            for (int r = 0; r < 8; ++r) {
                float v = sc[t][r] * ATT_SCALE;
                if (diag) {
                    int col = kt*128 + t*16 + l16;
                    int row = qt*128 + wave*16 + r + (hi ? 8 : 0);
                    if (col > row) v = -__builtin_inff();
                }
                sc[t][r] = v;
            }

        // online softmax per owned row (16-lane row reductions)
        for (int r = 0; r < 8; ++r) {
            float rm = sc[0][r];
            for (int t = 1; t < 8; ++t) rm = fmaxf(rm, sc[t][r]);
            for (int m = 8; m >= 1; m >>= 1) rm = fmaxf(rm, __shfl_xor(rm, m, 32));
            float mnew  = fmaxf(mrow[r], rm);
            float alpha = __expf(mrow[r] - mnew);
            float rs = 0.f;
            for (int t = 0; t < 8; ++t) {
                float p = __expf(sc[t][r] - mnew);
                sc[t][r] = p;
                rs += p;
            }
            for (int m = 8; m >= 1; m >>= 1) rs += __shfl_xor(rs, m, 32);
            lrow[r] = lrow[r] * alpha + rs;
            mrow[r] = mnew;
            for (int d = 0; d < 8; ++d) o[d][r] *= alpha;
        }

        // D-layout P -> A-layout bf16 via wave-private LDS
        unsigned short* pW = sP + wave * 2048;
        for (int t = 0; t < 8; ++t)
            for (int r = 0; r < 8; ++r)
                pW[(r + (hi ? 8 : 0)) * 128 + t*16 + l16] = f2bf(sc[t][r]);

        // O += P @ V : 4 key chunks x 8 dim subtiles
        for (int c = 0; c < 4; ++c) {
            Frag aF;
            aF.q[0] = *(const u32x4*)(pW + l16*128 + c*32 + kb);
            aF.q[1] = *(const u32x4*)(pW + l16*128 + c*32 + 16 + kb);
            for (int d = 0; d < 8; ++d) {
                Frag bF;
                int n = d*16 + l16;                 // dim column
                bF.q[0] = *(const u32x4*)(sV + n*128 + c*32 + kbB);
                bF.q[1] = *(const u32x4*)(sV + n*128 + c*32 + 16 + kbB);
                o[d] = __builtin_amdgcn_wmma_f32_16x16x32_bf16(
                        false, aF.v, false, bF.v, (short)0, o[d], false, false);
            }
        }
    }

    // epilogue: normalize, store bf16 [token, H*128]
    for (int r = 0; r < 8; ++r) {
        float inv = 1.0f / lrow[r];
        int row = qt*128 + wave*16 + r + (hi ? 8 : 0);
        unsigned short* op = O + ((long long)b * SEQ + row) * (NH*HD) + h*HD;
        for (int d = 0; d < 8; ++d)
            op[d*16 + l16] = f2bf(o[d][r] * inv);
    }
}

// =====================================================================
// host side
// =====================================================================
extern "C" void kernel_launch(void* const* d_in, const int* in_sizes, int n_in,
                              void* d_out, int out_size, void* d_ws, size_t ws_size,
                              hipStream_t stream) {
    const float* x  = (const float*)d_in[0];
    const float* fc = (const float*)d_in[1];
    const float* wq = (const float*)d_in[2];
    const float* wk = (const float*)d_in[3];
    const float* wv = (const float*)d_in[4];
    const float* wo = (const float*)d_in[5];
    const float* qw = (const float*)d_in[6];
    const float* kw = (const float*)d_in[7];

    char* ws = (char*)d_ws;
    size_t off = 0;
    auto alloc = [&](size_t bytes) -> void* {
        void* p = ws + off;
        off = (off + bytes + 255) & ~(size_t)255;
        return p;
    };

    unsigned short* xh  = (unsigned short*)alloc((size_t)MROWS * DMODEL * 2);
    unsigned short* wqT = (unsigned short*)alloc((size_t)DMODEL * 2048 * 2);
    unsigned short* wkT = (unsigned short*)alloc((size_t)DMODEL * 1024 * 2);
    unsigned short* wvT = (unsigned short*)alloc((size_t)DMODEL * 1024 * 2);
    unsigned short* woT = (unsigned short*)alloc((size_t)2048 * 2048 * 2);
    float* qf = (float*)alloc((size_t)MROWS * 2048 * 4);
    float* kf = (float*)alloc((size_t)MROWS * 1024 * 4);
    float* vf = (float*)alloc((size_t)MROWS * 1024 * 4);
    unsigned short* qbf = (unsigned short*)alloc((size_t)BB * NH   * SEQ * HD * 2);
    unsigned short* kbf = (unsigned short*)alloc((size_t)BB * NKVH * SEQ * HD * 2);
    unsigned short* vtb = (unsigned short*)alloc((size_t)BB * NKVH * HD * SEQ * 2);
    unsigned short* aob = (unsigned short*)alloc((size_t)MROWS * 2048 * 2);

    // 1. casts / packs
    cast_bf16<<<(MROWS * DMODEL) / 256, 256, 0, stream>>>(x, xh, (long long)MROWS * DMODEL);
    pack_wT<<<(2048 * 2048) / 256, 256, 0, stream>>>(wq, wqT, 2048, 2048);
    pack_wT<<<(1024 * 2048) / 256, 256, 0, stream>>>(wk, wkT, 1024, 2048);
    pack_wT<<<(1024 * 2048) / 256, 256, 0, stream>>>(wv, wvT, 1024, 2048);
    pack_wT<<<(2048 * 2048) / 256, 256, 0, stream>>>(wo, woT, 2048, 2048);

    // 2. QKV projections (bf16 WMMA GEMM)
    gemm_bf16<<<dim3(2048/128, MROWS/128), 256, 0, stream>>>(xh, wqT, qf, MROWS, 2048, DMODEL);
    gemm_bf16<<<dim3(1024/128, MROWS/128), 256, 0, stream>>>(xh, wkT, kf, MROWS, 1024, DMODEL);
    gemm_bf16<<<dim3(1024/128, MROWS/128), 256, 0, stream>>>(xh, wvT, vf, MROWS, 1024, DMODEL);

    // 3. RMSNorm + RoPE (Q, K) and V transpose
    norm_rope<<<BB * SEQ * NH,   64, 0, stream>>>(qf, fc, qw, qbf, NH);
    norm_rope<<<BB * SEQ * NKVH, 64, 0, stream>>>(kf, fc, kw, kbf, NKVH);
    v_trans<<<(BB * NKVH * HD * SEQ) / 256, 256, 0, stream>>>(vf, vtb);

    // 4. flash attention (96 KB dynamic LDS: K 32K + V 32K + P 32K)
    size_t smem = (size_t)(128*128 + 128*128 + 8*16*128) * 2;
    flash_attn<<<BB * NH * (SEQ/128), 256, smem, stream>>>(qbf, kbf, vtb, aob);

    // 5. output projection -> fp32 d_out
    gemm_bf16<<<dim3(2048/128, MROWS/128), 256, 0, stream>>>(aob, woT, (float*)d_out, MROWS, 2048, DMODEL);

    (void)in_sizes; (void)n_in; (void)out_size; (void)ws_size;
}
